// VFE_28037546508645
// MI455X (gfx1250) — matrix operand order repaired and checked by reading, not compiled, not run on hardware
//
#include <hip/hip_runtime.h>
#include <hip/hip_bf16.h>

// ---------------------------------------------------------------------------
// CDNA5 (gfx1250) implicit-GEMM 3D conv stack using V_WMMA.
// Wave32. Pair-mode: each wave computes a 16x32 (voxels x Cout) tile; the A
// fragment (gathered, expensive) is shared across two v_wmma_f32_16x16x32_f16
// issues. B fragments are staged through LDS with double-buffered
// global_load_async_to_lds_b128 (ASYNCcnt path), overlapped with the A gather.
// ---------------------------------------------------------------------------

typedef __attribute__((ext_vector_type(16))) _Float16 v16h;
typedef __attribute__((ext_vector_type(8)))  _Float16 v8h;
typedef __attribute__((ext_vector_type(8)))  float    v8f;

// Grid constants (match reference): PC range / 0.1 voxel
#define GB 2
#define GD 41
#define GH 128
#define GW 128
#define NPTS 200000
#define MVOX (GB*GD*GH*GW)

// ---------------------------------------------------------------- utilities
__global__ void zero_f32(float* __restrict__ p, int n) {
    int i = blockIdx.x * blockDim.x + threadIdx.x;
    if (i < n) p[i] = 0.0f;
}

// ------------------------------------------------------------- voxelization
__global__ void voxelize_scatter(const float* __restrict__ pts,
                                 float* __restrict__ feats,  // [MVOX][4]
                                 float* __restrict__ cnt,    // [MVOX]
                                 int n) {
    int i = blockIdx.x * blockDim.x + threadIdx.x;
    if (i >= n) return;
    float pb = pts[i*5+0];
    float px = pts[i*5+1];
    float py = pts[i*5+2];
    float pz = pts[i*5+3];
    float pi = pts[i*5+4];
    int b  = (int)pb;
    int ix = (int)floorf((px - 0.0f)  / 0.1f);
    int iy = (int)floorf((py + 6.4f)  / 0.1f);
    int iz = (int)floorf((pz + 3.0f)  / 0.1f);
    if (ix < 0 || iy < 0 || iz < 0 || ix >= GW || iy >= GH || iz >= (GD-1)) return;
    int lin = ((b * GD + iz) * GH + iy) * GW + ix;
    atomicAdd(&feats[(size_t)lin*4+0], px);
    atomicAdd(&feats[(size_t)lin*4+1], py);
    atomicAdd(&feats[(size_t)lin*4+2], pz);
    atomicAdd(&feats[(size_t)lin*4+3], pi);
    atomicAdd(&cnt[lin], 1.0f);
}

__global__ void voxelize_norm(const float* __restrict__ feats,
                              const float* __restrict__ cnt,
                              _Float16* __restrict__ act,   // [MVOX][4] f16
                              float* __restrict__ mask,     // [MVOX]
                              int mvox) {
    int v = blockIdx.x * blockDim.x + threadIdx.x;
    if (v >= mvox) return;
    float c = cnt[v];
    float d = fmaxf(c, 1.0f);
#pragma unroll
    for (int ch = 0; ch < 4; ++ch)
        act[(size_t)v*4+ch] = (_Float16)(feats[(size_t)v*4+ch] / d);
    mask[v] = (c > 0.0f) ? 1.0f : 0.0f;
}

// ---------------------------------------------------------- per-layer preps
__global__ void bn_prep(const float* __restrict__ g, const float* __restrict__ be,
                        const float* __restrict__ mu, const float* __restrict__ va,
                        float* __restrict__ bn, int C) {
    int i = blockIdx.x * blockDim.x + threadIdx.x;
    if (i >= C) return;
    float s = g[i] * rsqrtf(va[i] + 0.001f);
    bn[i]     = s;
    bn[C + i] = be[i] - mu[i] * s;
}

// Repack fp32 weights [K][Cout] (K = kd*kh*kw*Cin, cin fastest) into the
// WMMA B-matrix wave layout, f16:
//   wrep[((kChunk*nTiles + nTile)*32 + lane)*16 + j]
//   lane 0..15 : K = kBase + (j&7) + 16*(j>>3),     N = nTile*16 + lane
//   lane 16..31: K = kBase + 8 + (j&7) + 16*(j>>3), N = nTile*16 + (lane-16)
__global__ void repack_w(const float* __restrict__ w, _Float16* __restrict__ wrep,
                         int K, int Cout, int nTiles, int Kch) {
    int idx = blockIdx.x * blockDim.x + threadIdx.x;
    int total = Kch * nTiles * 32 * 16;
    if (idx >= total) return;
    int j    = idx & 15;
    int lane = (idx >> 4) & 31;
    int t    = idx >> 9;
    int nTile = t % nTiles;
    int kc    = t / nTiles;
    int k = (kc << 5) + ((lane >> 4) << 3) + (j & 7) + ((j >> 3) << 4);
    int n = (nTile << 4) + (lane & 15);
    _Float16 v = (_Float16)0.0f;
    if (k < K && n < Cout) v = (_Float16)w[(size_t)k * Cout + n];
    wrep[idx] = v;
}

// Mask dilation for spconv layers: out-mask = any(in-mask in receptive field)
__global__ void mask_update(const float* __restrict__ mi, float* __restrict__ mo,
                            int B, int Di, int Hi, int Wi, int Do, int Ho, int Wo,
                            int KD, int KH, int KW, int SD, int SH, int SW,
                            int PD, int PH, int PW) {
    int m = blockIdx.x * blockDim.x + threadIdx.x;
    int Mtot = B * Do * Ho * Wo;
    if (m >= Mtot) return;
    int xo = m % Wo; int t = m / Wo;
    int yo = t % Ho; t /= Ho;
    int zo = t % Do; int b = t / Do;
    float any = 0.0f;
    for (int kd = 0; kd < KD; ++kd) {
        int zi = zo * SD - PD + kd; if (zi < 0 || zi >= Di) continue;
        for (int kh = 0; kh < KH; ++kh) {
            int yi = yo * SH - PH + kh; if (yi < 0 || yi >= Hi) continue;
            for (int kw = 0; kw < KW; ++kw) {
                int xi = xo * SW - PW + kw; if (xi < 0 || xi >= Wi) continue;
                if (mi[(((size_t)b * Di + zi) * Hi + yi) * Wi + xi] > 0.0f) any = 1.0f;
            }
        }
    }
    mo[m] = any;
}

// ------------------------------------------------------------- conv (WMMA)
struct ConvP {
    const _Float16* __restrict__ x;     // [B,Di,Hi,Wi,Cin] f16
    const _Float16* __restrict__ wrep;  // repacked B-layout weights
    const float*    __restrict__ bn;    // [Cout] scale, [Cout] bias
    const float*    __restrict__ mask;  // [B,Do,Ho,Wo]
    _Float16*       __restrict__ y;     // [B,Do,Ho,Wo,Cout] f16
    int Di, Hi, Wi, Do, Ho, Wo;
    int Cin, Cout, cinShift;
    int KD, KH, KW, SD, SH, SW, PD, PH, PW;
    int K, Kch;           // K total, number of K/32 chunks
    int Mtot, nTiles, totTiles;
};

// Issue async copy of this lane's 32B B-fragment into its LDS slot.
// Per CDNA5 ISA: the instruction offset is added to both the LDS and the global address
__device__ __forceinline__ void async_copy_b32B(unsigned ldsAddr, const _Float16* src) {
    unsigned long long ga = (unsigned long long)(uintptr_t)src;
    asm volatile("global_load_async_to_lds_b128 %0, %1, off"
                 :: "v"(ldsAddr), "v"(ga) : "memory");
    asm volatile("global_load_async_to_lds_b128 %0, %1, off offset:16"
                 :: "v"(ldsAddr), "v"(ga) : "memory");
}

__device__ __forceinline__ void wait_async0() {
    asm volatile("s_wait_asynccnt 0x0" ::: "memory");
}
__device__ __forceinline__ void wait_async4() {
    asm volatile("s_wait_asynccnt 0x4" ::: "memory");
}

__global__ __launch_bounds__(256) void conv_wmma(ConvP p) {
    // [wave][buf][nPairSlot][lane][16 halves] : 32KB
    __shared__ _Float16 ldsB[8][2][2][32][16];

    const int lane = threadIdx.x & 31;
    const int wv   = (threadIdx.x >> 5) & 7;
    const int row  = lane & 15;          // M row (A) / N col (B,C,D)
    const int hi   = lane >> 4;          // lane half select
    const int wavesTotal = (gridDim.x * blockDim.x) >> 5;
    int wave = (blockIdx.x * blockDim.x + threadIdx.x) >> 5;

    const int cinMask = (1 << p.cinShift) - 1;
    const int wStride = p.nTiles * 32 * 16;   // halves between K-chunks in wrep
    const int kOff = hi << 3;                 // 0 for lanes 0..15, 8 for lanes 16..31

    const bool pairMode = ((p.nTiles & 1) == 0) && (p.cinShift >= 4);
    const int nUnits = pairMode ? (p.nTiles >> 1) : p.nTiles;

    unsigned ldsBase = (unsigned)(uintptr_t)&ldsB[wv][0][0][lane][0];

    for (int tile = wave; tile < p.totTiles; tile += wavesTotal) {
        int mTile = tile / nUnits;
        int nUnit = tile - mTile * nUnits;
        int mBase = mTile << 4;

        // Decode this lane's A row (output voxel) -> input base coords
        int m  = mBase + row;
        int mc = (m < p.Mtot) ? m : (p.Mtot - 1);
        int xo = mc % p.Wo; int t = mc / p.Wo;
        int yo = t % p.Ho;  t /= p.Ho;
        int zo = t % p.Do;  int b = t / p.Do;
        int zi0 = zo * p.SD - p.PD;
        int yi0 = yo * p.SH - p.PH;
        int xi0 = xo * p.SW - p.PW;

        if (pairMode) {
            // ================= 16x32 tile: one A, two WMMAs =================
            int nTile0 = nUnit << 1;
            const _Float16* wr0 = p.wrep + ((size_t)nTile0 * 32 + lane) * 16;
            const _Float16* wr1 = wr0 + 32 * 16;
            v8f acc0 = {}, acc1 = {};

            // prologue: stage chunk 0 into buffer 0
            async_copy_b32B(ldsBase,        wr0);
            async_copy_b32B(ldsBase + 1024, wr1);

            for (int kc = 0; kc < p.Kch; ++kc) {
                int cur = kc & 1;
                bool hasNext = (kc + 1) < p.Kch;
                if (hasNext) {   // double-buffer: stage next chunk now
                    unsigned nb = ldsBase + (cur ^ 1) * 2048;
                    async_copy_b32B(nb,        wr0 + (size_t)(kc + 1) * wStride);
                    async_copy_b32B(nb + 1024, wr1 + (size_t)(kc + 1) * wStride);
                }

                // ---- gather A: two contiguous b128 segments per lane ----
                int kBase = kc << 5;
                v16h a;
#pragma unroll
                for (int s = 0; s < 2; ++s) {
                    int k0 = kBase + kOff + (s << 4);
                    v8h seg = {};
                    if (k0 < p.K) {
                        int tap  = k0 >> p.cinShift;
                        int cin0 = k0 & cinMask;
                        int kwx = tap % p.KW; int r2 = tap / p.KW;
                        int khy = r2 % p.KH; int kdz = r2 / p.KH;
                        int zi = zi0 + kdz, yi = yi0 + khy, xi = xi0 + kwx;
                        if (zi >= 0 && zi < p.Di && yi >= 0 && yi < p.Hi &&
                            xi >= 0 && xi < p.Wi) {
                            size_t off = ((((size_t)b * p.Di + zi) * p.Hi + yi) * p.Wi + xi)
                                             * p.Cin + cin0;
                            seg = *(const v8h*)(p.x + off);
                        }
                    }
#pragma unroll
                    for (int j = 0; j < 8; ++j) a[(s << 3) + j] = seg[j];
                }

                // in-order ASYNCcnt: <=4 outstanding means current chunk landed
                if (hasNext) wait_async4(); else wait_async0();

                v16h bm0 = *(const v16h*)&ldsB[wv][cur][0][lane][0];
                v16h bm1 = *(const v16h*)&ldsB[wv][cur][1][lane][0];
                acc0 = __builtin_amdgcn_wmma_f32_16x16x32_f16(
                    false, a, false, bm0, (short)0, acc0, false, false);
                acc1 = __builtin_amdgcn_wmma_f32_16x16x32_f16(
                    false, a, false, bm1, (short)0, acc1, false, false);
            }

            // ---- epilogue: BN + ReLU + mask, store f16 (two n-tiles) ----
            int n0 = (nTile0 << 4) + row;
            float s0 = p.bn[n0],      b0 = p.bn[p.Cout + n0];
            float s1 = p.bn[n0 + 16], b1 = p.bn[p.Cout + n0 + 16];
#pragma unroll
            for (int r = 0; r < 8; ++r) {
                int mo = mBase + r + (hi << 3);
                if (mo < p.Mtot) {
                    float mk = p.mask[mo];
                    float v0 = fmaf(acc0[r], s0, b0);
                    v0 = (v0 > 0.0f ? v0 : 0.0f) * mk;
                    float v1 = fmaf(acc1[r], s1, b1);
                    v1 = (v1 > 0.0f ? v1 : 0.0f) * mk;
                    p.y[(size_t)mo * p.Cout + n0]      = (_Float16)v0;
                    p.y[(size_t)mo * p.Cout + n0 + 16] = (_Float16)v1;
                }
            }
        } else {
            // ================= single 16x16 tile =================
            int nTile = nUnit;
            const _Float16* wr = p.wrep + ((size_t)nTile * 32 + lane) * 16;
            v8f acc = {};

            if (p.cinShift >= 4) {
                for (int kc = 0; kc < p.Kch; ++kc) {
                    int kBase = kc << 5;
                    async_copy_b32B(ldsBase, wr + (size_t)kc * wStride);

                    v16h a;
#pragma unroll
                    for (int s = 0; s < 2; ++s) {
                        int k0 = kBase + kOff + (s << 4);
                        v8h seg = {};
                        if (k0 < p.K) {
                            int tap  = k0 >> p.cinShift;
                            int cin0 = k0 & cinMask;
                            int kwx = tap % p.KW; int r2 = tap / p.KW;
                            int khy = r2 % p.KH; int kdz = r2 / p.KH;
                            int zi = zi0 + kdz, yi = yi0 + khy, xi = xi0 + kwx;
                            if (zi >= 0 && zi < p.Di && yi >= 0 && yi < p.Hi &&
                                xi >= 0 && xi < p.Wi) {
                                size_t off = ((((size_t)b * p.Di + zi) * p.Hi + yi) * p.Wi + xi)
                                                 * p.Cin + cin0;
                                seg = *(const v8h*)(p.x + off);
                            }
                        }
#pragma unroll
                        for (int j = 0; j < 8; ++j) a[(s << 3) + j] = seg[j];
                    }

                    wait_async0();
                    v16h bm = *(const v16h*)&ldsB[wv][0][0][lane][0];
                    acc = __builtin_amdgcn_wmma_f32_16x16x32_f16(
                        false, a, false, bm, (short)0, acc, false, false);
                }
            } else {
                // generic scalar gather (layer 0, Cin = 4)
                for (int kc = 0; kc < p.Kch; ++kc) {
                    int kBase = kc << 5;
                    async_copy_b32B(ldsBase, wr + (size_t)kc * wStride);

                    v16h a = {};
#pragma unroll
                    for (int j = 0; j < 16; ++j) {
                        int k = kBase + kOff + (j & 7) + ((j >> 3) << 4);
                        _Float16 v = (_Float16)0.0f;
                        if (k < p.K) {
                            int cin = k & cinMask;
                            int kv  = k >> p.cinShift;
                            int kwx = kv % p.KW; int r2 = kv / p.KW;
                            int khy = r2 % p.KH; int kdz = r2 / p.KH;
                            int zi = zi0 + kdz, yi = yi0 + khy, xi = xi0 + kwx;
                            if (zi >= 0 && zi < p.Di && yi >= 0 && yi < p.Hi &&
                                xi >= 0 && xi < p.Wi) {
                                size_t off = ((((size_t)b * p.Di + zi) * p.Hi + yi) * p.Wi + xi)
                                                 * p.Cin + cin;
                                v = p.x[off];
                            }
                        }
                        a[j] = v;
                    }

                    wait_async0();
                    v16h bm = *(const v16h*)&ldsB[wv][0][0][lane][0];
                    acc = __builtin_amdgcn_wmma_f32_16x16x32_f16(
                        false, a, false, bm, (short)0, acc, false, false);
                }
            }

            int n = (nTile << 4) + row;
            float s   = p.bn[n];
            float bia = p.bn[p.Cout + n];
#pragma unroll
            for (int r = 0; r < 8; ++r) {
                int mo = mBase + r + (hi << 3);
                if (mo < p.Mtot) {
                    float val = fmaf(acc[r], s, bia);
                    val = val > 0.0f ? val : 0.0f;
                    val *= p.mask[mo];
                    p.y[(size_t)mo * p.Cout + n] = (_Float16)val;
                }
            }
        }
    }
}

// --------------------------------------------------------------- final copy
// act: [2][2][16][16][128] f16  ->  out: [2][256][16][16] f32,
// out[b][c*2+d][h][w] = act[b][d][h][w][c]
__global__ void finalize_out(const _Float16* __restrict__ act, float* __restrict__ out) {
    int i = blockIdx.x * blockDim.x + threadIdx.x;
    if (i >= 2*256*16*16) return;
    int w = i & 15; int t = i >> 4;
    int h = t & 15; t >>= 4;
    int cd = t & 255; int b = t >> 8;
    int d = cd & 1; int c = cd >> 1;
    out[i] = (float)act[((((size_t)b*2 + d)*16 + h)*16 + w)*128 + c];
}

// ------------------------------------------------------------------- driver
struct LayerCfg {
    int Cin, Cout, KD, KH, KW, SD, SH, SW, PD, PH, PW;
    int Di, Hi, Wi, Do, Ho, Wo, sp;
};

extern "C" void kernel_launch(void* const* d_in, const int* in_sizes, int n_in,
                              void* d_out, int out_size, void* d_ws, size_t ws_size,
                              hipStream_t stream) {
    (void)in_sizes; (void)n_in; (void)out_size; (void)ws_size;

    static const LayerCfg T[12] = {
        { 4, 16, 3,3,3, 1,1,1, 1,1,1, 41,128,128, 41,128,128, 0},
        {16, 16, 3,3,3, 1,1,1, 1,1,1, 41,128,128, 41,128,128, 0},
        {16, 32, 3,3,3, 2,2,2, 1,1,1, 41,128,128, 21, 64, 64, 1},
        {32, 32, 3,3,3, 1,1,1, 1,1,1, 21, 64, 64, 21, 64, 64, 0},
        {32, 32, 3,3,3, 1,1,1, 1,1,1, 21, 64, 64, 21, 64, 64, 0},
        {32, 64, 3,3,3, 2,2,2, 1,1,1, 21, 64, 64, 11, 32, 32, 1},
        {64, 64, 3,3,3, 1,1,1, 1,1,1, 11, 32, 32, 11, 32, 32, 0},
        {64, 64, 3,3,3, 1,1,1, 1,1,1, 11, 32, 32, 11, 32, 32, 0},
        {64, 64, 3,3,3, 2,2,2, 0,1,1, 11, 32, 32,  5, 16, 16, 1},
        {64, 64, 3,3,3, 1,1,1, 1,1,1,  5, 16, 16,  5, 16, 16, 0},
        {64, 64, 3,3,3, 1,1,1, 1,1,1,  5, 16, 16,  5, 16, 16, 0},
        {64,128, 3,1,1, 2,1,1, 0,0,0,  5, 16, 16,  2, 16, 16, 1},
    };

    const float* points = (const float*)d_in[0];

    // ---- workspace carve-up (all 256B aligned) ----
    char* ws = (char*)d_ws;
    const size_t actBytes  = (size_t)MVOX * 16 * sizeof(_Float16); // 42,991,616
    const size_t maskBytes = (size_t)MVOX * sizeof(float);         //  5,373,952
    _Float16* actA = (_Float16*)(ws);
    _Float16* actB = (_Float16*)(ws + actBytes);
    float*    maskA = (float*)(ws + 2*actBytes);
    float*    maskB = (float*)(ws + 2*actBytes + maskBytes);
    _Float16* wrep  = (_Float16*)(ws + 2*actBytes + 2*maskBytes);
    float*    bn    = (float*)(ws + 2*actBytes + 2*maskBytes + 256*1024);

    // voxelize accumulators alias actB (dead until conv0 writes it)
    float* feats = (float*)actB;
    float* cnt   = feats + (size_t)MVOX * 4;

    // ---- voxelize ----
    int nz = MVOX * 5;
    zero_f32<<<(nz + 255)/256, 256, 0, stream>>>(feats, nz);
    voxelize_scatter<<<(NPTS + 255)/256, 256, 0, stream>>>(points, feats, cnt, NPTS);
    voxelize_norm<<<(MVOX + 255)/256, 256, 0, stream>>>(feats, cnt, actA, maskA, MVOX);

    // ---- conv stack ----
    _Float16* xin = actA; _Float16* xout = actB;
    float* mcur = maskA; float* mnext = maskB;

    for (int i = 0; i < 12; ++i) {
        const LayerCfg& l = T[i];
        const float* w  = (const float*)d_in[1 + 5*i + 0];
        const float* g  = (const float*)d_in[1 + 5*i + 1];
        const float* be = (const float*)d_in[1 + 5*i + 2];
        const float* mu = (const float*)d_in[1 + 5*i + 3];
        const float* va = (const float*)d_in[1 + 5*i + 4];

        bn_prep<<<1, 128, 0, stream>>>(g, be, mu, va, bn, l.Cout);

        int K   = l.KD * l.KH * l.KW * l.Cin;
        int Kch = (K + 31) / 32;
        int nT  = l.Cout / 16;
        int repTot = Kch * nT * 32 * 16;
        repack_w<<<(repTot + 255)/256, 256, 0, stream>>>(w, wrep, K, l.Cout, nT, Kch);

        int Mtot = GB * l.Do * l.Ho * l.Wo;
        if (l.sp) {
            mask_update<<<(Mtot + 255)/256, 256, 0, stream>>>(
                mcur, mnext, GB, l.Di, l.Hi, l.Wi, l.Do, l.Ho, l.Wo,
                l.KD, l.KH, l.KW, l.SD, l.SH, l.SW, l.PD, l.PH, l.PW);
            float* tm = mcur; mcur = mnext; mnext = tm;
        }

        int cinShift = (l.Cin == 4) ? 2 : (l.Cin == 16) ? 4 : (l.Cin == 32) ? 5 : 6;
        bool pairMode = ((nT & 1) == 0) && (cinShift >= 4);
        int nUnits = pairMode ? (nT >> 1) : nT;

        ConvP p;
        p.x = xin; p.wrep = wrep; p.bn = bn; p.mask = mcur; p.y = xout;
        p.Di = l.Di; p.Hi = l.Hi; p.Wi = l.Wi;
        p.Do = l.Do; p.Ho = l.Ho; p.Wo = l.Wo;
        p.Cin = l.Cin; p.Cout = l.Cout;
        p.cinShift = cinShift;
        p.KD = l.KD; p.KH = l.KH; p.KW = l.KW;
        p.SD = l.SD; p.SH = l.SH; p.SW = l.SW;
        p.PD = l.PD; p.PH = l.PH; p.PW = l.PW;
        p.K = K; p.Kch = Kch;
        p.Mtot = Mtot; p.nTiles = nT;
        int mT = (Mtot + 15) / 16;
        p.totTiles = mT * nUnits;

        int blocks = (p.totTiles + 7) / 8;   // 8 waves per 256-thread block
        conv_wmma<<<blocks, 256, 0, stream>>>(p);

        _Float16* tx = xin; xin = xout; xout = tx;
    }

    finalize_out<<<(2*256*16*16 + 255)/256, 256, 0, stream>>>(xin, (float*)d_out);
}
